// MultiQueryAttention_61383672594618
// MI455X (gfx1250) — matrix operand order repaired and checked by reading, not compiled, not run on hardware
//
#include <hip/hip_runtime.h>

typedef __bf16 bf16_t;
typedef bf16_t v16bf __attribute__((ext_vector_type(16)));
typedef float  v8f   __attribute__((ext_vector_type(8)));
typedef unsigned short u16x8 __attribute__((ext_vector_type(8)));
typedef unsigned short u16x4 __attribute__((ext_vector_type(4)));
typedef float  f32x4 __attribute__((ext_vector_type(4)));
typedef int    i32x4v __attribute__((vector_size(16)));   // matches builtin param type

union FragU { v16bf bf; u16x8 h[2]; };

#if defined(__HIP_DEVICE_COMPILE__) && __has_builtin(__builtin_amdgcn_global_load_async_to_lds_b128)
#define HAS_ASYNC_LDS 1
#else
#define HAS_ASYNC_LDS 0
#endif

__device__ __forceinline__ void wait_async_all() {
#if defined(__HIP_DEVICE_COMPILE__)
#if __has_builtin(__builtin_amdgcn_s_wait_asynccnt)
    __builtin_amdgcn_s_wait_asynccnt(0);
#else
    asm volatile("s_wait_asynccnt 0x0" ::: "memory");
#endif
#endif
}

// 16-byte global -> LDS copy; async DMA (ASYNCcnt) when available.
__device__ __forceinline__ void copy16_g2l(const unsigned short* __restrict__ g,
                                           unsigned short* l) {
#if HAS_ASYNC_LDS
    __builtin_amdgcn_global_load_async_to_lds_b128(
        (__attribute__((address_space(1))) i32x4v*)g,
        (__attribute__((address_space(3))) i32x4v*)l, 0, 0);
#else
    *(u16x8*)l = *(const u16x8*)g;
#endif
}

__device__ __forceinline__ unsigned short f32_to_bf16_rne(float f) {
    union { float f; unsigned int u; } c; c.f = f;
    unsigned int u = c.u;
    return (unsigned short)((u + 0x7FFFu + ((u >> 16) & 1u)) >> 16);
}

// ---------------------------------------------------------------------------
// fp32 -> bf16 bulk convert (4 elements / thread, 16B loads)
// ---------------------------------------------------------------------------
__global__ __launch_bounds__(256)
void cvt_f32_bf16(const float* __restrict__ in, unsigned short* __restrict__ out,
                  long long n) {
    long long i = ((long long)blockIdx.x * blockDim.x + threadIdx.x) * 4;
    if (i + 3 < n) {
        f32x4 v = *(const f32x4*)(in + i);
        u16x4 h;
        #pragma unroll
        for (int e = 0; e < 4; ++e) h[e] = f32_to_bf16_rne(v[e]);
        *(u16x4*)(out + i) = h;
    }
}

// ---------------------------------------------------------------------------
// Generic batched WMMA GEMM:  C = alpha * (A @ B + bias)
//   A: [M,K]  (bf16, or f32 converted on LDS fill if A_F32)
//   B: [K,N] bf16 row-major, or [N,K] row-major if B_TRANS (i.e. C = A @ B^T)
//   Per-z base offset = (z/zdiv)*stride_hi + (z%zdiv)*stride_lo  (elements)
//   Output f32, or bf16 if OUT_BF16.
// Tiling: BMxBN block tile, BK=32, WAVES_M x WAVES_N waves of 32 lanes,
// each wave computes FMxFN 16x16 f32 accumulators via v_wmma_f32_16x16x32_bf16.
// Double-buffered LDS; straight-copy tiles use GLOBAL_LOAD_ASYNC_TO_LDS_B128.
// ---------------------------------------------------------------------------
template<int BM, int BN, int WAVES_M, int WAVES_N, bool B_TRANS, bool A_F32, bool OUT_BF16>
__global__ __launch_bounds__(WAVES_M * WAVES_N * 32)
void wmma_gemm(const void* __restrict__ Ag, long long lda, long long sAhi, long long sAlo,
               const unsigned short* __restrict__ Bg, long long ldb, long long sBhi, long long sBlo,
               void* __restrict__ Cg, long long ldc, long long sChi, long long sClo,
               const float* __restrict__ bias, float alpha, int zdiv, int K)
{
    constexpr int BK = 32;
    constexpr int LS = BK + 8;                        // padded LDS stride (u16)
    constexpr int THREADS = WAVES_M * WAVES_N * 32;
    constexpr int FM = BM / (WAVES_M * 16);
    constexpr int FN = BN / (WAVES_N * 16);
    constexpr bool ASYNC_USED = (!A_F32) || B_TRANS;

    __shared__ unsigned short sA[2][BM * LS];
    __shared__ unsigned short sB[2][BN * LS];

    const int tid  = threadIdx.x;
    const int lane = tid & 31;
    const int wave = tid >> 5;
    const int wm   = wave / WAVES_N;
    const int wn   = wave % WAVES_N;
    const int half = lane >> 4;      // 0: lanes 0-15, 1: lanes 16-31
    const int lr   = lane & 15;

    const int z = blockIdx.z;
    const long long zhi = (long long)(z / zdiv);
    const long long zlo = (long long)(z % zdiv);

    const unsigned short* Bp = Bg + zhi * sBhi + zlo * sBlo;
    const float*          Af = nullptr;
    const unsigned short* Ah = nullptr;
    if constexpr (A_F32) Af = (const float*)Ag + zhi * sAhi + zlo * sAlo;
    else                 Ah = (const unsigned short*)Ag + zhi * sAhi + zlo * sAlo;

    const long long blockM = (long long)blockIdx.y * BM;
    const long long blockN = (long long)blockIdx.x * BN;

    // ---- tile loaders (issue only; completion via asynccnt/dscnt + barrier) ----
    auto loadA = [&](int buf, int kt) {
        constexpr int ITER = (BM * BK / 8) / THREADS;
        #pragma unroll
        for (int t = 0; t < ITER; ++t) {
            int c = tid + t * THREADS;
            int i = c >> 2;            // BK/8 = 4 chunks per row
            int j = (c & 3) * 8;
            if constexpr (A_F32) {
                const float* src = Af + (blockM + i) * lda + kt + j;
                f32x4 a0 = *(const f32x4*)(src);
                f32x4 a1 = *(const f32x4*)(src + 4);
                u16x8 h;
                #pragma unroll
                for (int e = 0; e < 4; ++e) h[e]     = f32_to_bf16_rne(a0[e]);
                #pragma unroll
                for (int e = 0; e < 4; ++e) h[e + 4] = f32_to_bf16_rne(a1[e]);
                *(u16x8*)&sA[buf][i * LS + j] = h;
            } else {
                copy16_g2l(Ah + (blockM + i) * lda + kt + j, &sA[buf][i * LS + j]);
            }
        }
    };
    auto loadB = [&](int buf, int kt) {
        if constexpr (B_TRANS) {
            // B stored [N,K] row-major: straight 16B async copies
            constexpr int ITER = (BN * BK / 8) / THREADS;
            #pragma unroll
            for (int t = 0; t < ITER; ++t) {
                int c = tid + t * THREADS;
                int i = c >> 2;
                int j = (c & 3) * 8;
                copy16_g2l(Bp + (blockN + i) * ldb + kt + j, &sB[buf][i * LS + j]);
            }
        } else {
            // B stored [K,N] row-major: 16B load, 2B scatter-transpose
            constexpr int ITER = (BK * (BN / 8)) / THREADS;
            #pragma unroll
            for (int t = 0; t < ITER; ++t) {
                int c  = tid + t * THREADS;
                int k  = c / (BN / 8);
                int n0 = (c % (BN / 8)) * 8;
                u16x8 v = *(const u16x8*)(Bp + (long long)(kt + k) * ldb + blockN + n0);
                #pragma unroll
                for (int e = 0; e < 8; ++e) sB[buf][(n0 + e) * LS + k] = v[e];
            }
        }
    };

    v8f acc[FM][FN];
    #pragma unroll
    for (int i = 0; i < FM; ++i)
        #pragma unroll
        for (int j = 0; j < FN; ++j)
            #pragma unroll
            for (int e = 0; e < 8; ++e) acc[i][j][e] = 0.0f;

    const int steps = K / BK;
    loadA(0, 0);
    loadB(0, 0);

    for (int it = 0; it < steps; ++it) {
        if constexpr (ASYNC_USED) wait_async_all();
        __syncthreads();                       // tile `it` published to all waves

        if (it + 1 < steps) {                  // prefetch next tile into other buffer
            loadA((it + 1) & 1, (it + 1) * BK);
            loadB((it + 1) & 1, (it + 1) * BK);
        }
        const int buf = it & 1;

        // ---- fragments ----
        FragU afrag[FM], bfrag[FN];
        #pragma unroll
        for (int fm = 0; fm < FM; ++fm) {
            int row = wm * FM * 16 + fm * 16 + lr;
            const unsigned short* p = &sA[buf][row * LS + half * 8];
            afrag[fm].h[0] = *(const u16x8*)(p);        // K = abase .. abase+7
            afrag[fm].h[1] = *(const u16x8*)(p + 16);   // K = abase+16 .. abase+23
        }
        #pragma unroll
        for (int fn = 0; fn < FN; ++fn) {
            int col = wn * FN * 16 + fn * 16 + lr;
            const unsigned short* p = &sB[buf][col * LS + half * 8];
            bfrag[fn].h[0] = *(const u16x8*)(p);        // K = kbase .. kbase+7
            bfrag[fn].h[1] = *(const u16x8*)(p + 16);   // K = kbase+16 .. kbase+23
        }

        // ---- 16x16x32 bf16 WMMAs ----
        #pragma unroll
        for (int fm = 0; fm < FM; ++fm)
            #pragma unroll
            for (int fn = 0; fn < FN; ++fn)
                acc[fm][fn] = __builtin_amdgcn_wmma_f32_16x16x32_bf16(
                    false, afrag[fm].bf, false, bfrag[fn].bf,
                    (short)0, acc[fm][fn], false, false);
    }

    // ---- epilogue: alpha * (acc + bias) ----
    float*          Cf = nullptr;
    unsigned short* Cb = nullptr;
    if constexpr (OUT_BF16) Cb = (unsigned short*)Cg + zhi * sChi + zlo * sClo;
    else                    Cf = (float*)Cg + zhi * sChi + zlo * sClo;

    #pragma unroll
    for (int fm = 0; fm < FM; ++fm) {
        #pragma unroll
        for (int fn = 0; fn < FN; ++fn) {
            long long m0 = blockM + wm * FM * 16 + fm * 16 + half * 8;
            long long n  = blockN + wn * FN * 16 + fn * 16 + lr;
            float bb = bias ? bias[n] : 0.0f;
            #pragma unroll
            for (int r = 0; r < 8; ++r) {
                float v = alpha * (acc[fm][fn][r] + bb);
                if constexpr (OUT_BF16) Cb[(m0 + r) * ldc + n] = f32_to_bf16_rne(v);
                else                    Cf[(m0 + r) * ldc + n] = v;
            }
        }
    }
}

// ---------------------------------------------------------------------------
// Row softmax, fixed 2048 columns, one 256-thread block (8 waves) per row.
// ---------------------------------------------------------------------------
__global__ __launch_bounds__(256)
void softmax_rows_2048(float* __restrict__ p) {
    const long long row = blockIdx.x;
    float* r = p + row * 2048ll;
    const int tid  = threadIdx.x;
    const int wave = tid >> 5;
    const int lane = tid & 31;

    float v[8];
    float mx = -3.4e38f;
    #pragma unroll
    for (int i = 0; i < 8; ++i) { v[i] = r[tid + i * 256]; mx = fmaxf(mx, v[i]); }
    #pragma unroll
    for (int o = 16; o > 0; o >>= 1) mx = fmaxf(mx, __shfl_xor(mx, o, 32));

    __shared__ float red[8];
    if (lane == 0) red[wave] = mx;
    __syncthreads();
    float m = red[0];
    #pragma unroll
    for (int i = 1; i < 8; ++i) m = fmaxf(m, red[i]);

    float s = 0.0f;
    #pragma unroll
    for (int i = 0; i < 8; ++i) { v[i] = expf(v[i] - m); s += v[i]; }
    #pragma unroll
    for (int o = 16; o > 0; o >>= 1) s += __shfl_xor(s, o, 32);
    __syncthreads();
    if (lane == 0) red[wave] = s;
    __syncthreads();
    float tot = 0.0f;
    #pragma unroll
    for (int i = 0; i < 8; ++i) tot += red[i];
    float inv = 1.0f / tot;
    #pragma unroll
    for (int i = 0; i < 8; ++i) r[tid + i * 256] = v[i] * inv;
}

// ---------------------------------------------------------------------------
extern "C" void kernel_launch(void* const* d_in, const int* in_sizes, int n_in,
                              void* d_out, int out_size, void* d_ws, size_t ws_size,
                              hipStream_t stream) {
    (void)in_sizes; (void)n_in; (void)out_size; (void)ws_size;

    const float* query = (const float*)d_in[0];
    const float* key_  = (const float*)d_in[1];
    const float* value = (const float*)d_in[2];
    const float* q_W   = (const float*)d_in[3];
    const float* q_b   = (const float*)d_in[4];
    const float* k_W   = (const float*)d_in[5];
    const float* k_b   = (const float*)d_in[6];
    const float* v_W   = (const float*)d_in[7];
    const float* v_b   = (const float*)d_in[8];
    const float* o_W   = (const float*)d_in[9];
    const float* o_b   = (const float*)d_in[10];

    constexpr long long Bn = 2, S = 2048, DM = 1024, Hh = 16, DK = 64;
    constexpr long long MALL = Bn * S;          // 4096 rows
    constexpr long long OUT0 = Bn * S * DM;     // 4,194,304 floats (output part)

    // workspace carve-up (bf16 buffers)
    char* w = (char*)d_ws;
    auto take = [&](long long elems) -> unsigned short* {
        unsigned short* p = (unsigned short*)w;
        w += ((elems * 2 + 255) & ~255ll);
        return p;
    };
    unsigned short* qin = take(MALL * DM);   // query bf16
    unsigned short* kin = take(MALL * DM);   // key bf16
    unsigned short* vin = take(MALL * DM);   // value bf16
    unsigned short* qWb = take(DM * DM);
    unsigned short* kWb = take(DM * DK);
    unsigned short* vWb = take(DM * DK);
    unsigned short* oWb = take(DM * DM);
    unsigned short* Qp  = take(MALL * DM);   // scaled projected q, [B*S, H*DK]
    unsigned short* Kp  = take(MALL * DK);   // projected k, [B*S, DK]
    unsigned short* Vp  = take(MALL * DK);   // projected v, [B*S, DK]
    unsigned short* Xb  = take(MALL * DM);   // attention result, [B*S, H*DK]

    auto cvt = [&](const float* src, unsigned short* dst, long long n) {
        unsigned blocks = (unsigned)((n / 4 + 255) / 256);
        cvt_f32_bf16<<<dim3(blocks), dim3(256), 0, stream>>>(src, dst, n);
    };
    cvt(query, qin, MALL * DM);
    cvt(key_,  kin, MALL * DM);
    cvt(value, vin, MALL * DM);
    cvt(q_W,   qWb, DM * DM);
    cvt(k_W,   kWb, DM * DK);
    cvt(v_W,   vWb, DM * DK);
    cvt(o_W,   oWb, DM * DM);

    float* out   = (float*)d_out;
    float* pattn = out + OUT0;

    // Q projection (attention scale 1/sqrt(64)=0.125 folded in), bf16 out
    wmma_gemm<128,128,2,4,false,false,true>
        <<<dim3(DM/128, MALL/128, 1), 256, 0, stream>>>(
        qin, DM, 0, 0,  qWb, DM, 0, 0,  Qp, DM, 0, 0,  q_b, 0.125f, 1, (int)DM);

    // K / V projections (N=64), bf16 out
    wmma_gemm<128,64,4,2,false,false,true>
        <<<dim3(1, MALL/128, 1), 256, 0, stream>>>(
        kin, DM, 0, 0,  kWb, DK, 0, 0,  Kp, DK, 0, 0,  k_b, 1.0f, 1, (int)DM);
    wmma_gemm<128,64,4,2,false,false,true>
        <<<dim3(1, MALL/128, 1), 256, 0, stream>>>(
        vin, DM, 0, 0,  vWb, DK, 0, 0,  Vp, DK, 0, 0,  v_b, 1.0f, 1, (int)DM);

    // scores = Qh @ K^T  per (b,h); z = b*16+h; f32 straight into p_attn region
    wmma_gemm<128,128,2,4,true,false,false>
        <<<dim3(S/128, S/128, Bn*Hh), 256, 0, stream>>>(
        Qp,    DM, S*DM,    DK,
        Kp,    DK, S*DK,    0,
        pattn, S,  Hh*S*S,  S*S,
        nullptr, 1.0f, (int)Hh, (int)DK);

    // softmax in place over p_attn rows
    softmax_rows_2048<<<dim3((unsigned)(Bn*Hh*S)), 256, 0, stream>>>(pattn);

    // x = P @ V  per (b,h); A is f32 p_attn converted to bf16 on LDS fill
    wmma_gemm<128,64,4,2,false,true,true>
        <<<dim3(1, S/128, Bn*Hh), 256, 0, stream>>>(
        pattn, S,  Hh*S*S, S*S,
        Vp,    DK, S*DK,   0,
        Xb,    DM, S*DM,   DK,
        nullptr, 1.0f, (int)Hh, (int)S);

    // output projection -> f32 d_out[0 .. OUT0)
    wmma_gemm<128,128,2,4,false,false,false>
        <<<dim3(DM/128, MALL/128, 1), 256, 0, stream>>>(
        Xb, DM, 0, 0,  oWb, DM, 0, 0,  out, DM, 0, 0,  o_b, 1.0f, 1, (int)DM);
}
